// GraphSAGEEncoder_83485574299694
// MI455X (gfx1250) — matrix operand order repaired and checked by reading, not compiled, not run on hardware
//
#include <hip/hip_runtime.h>
#include <hip/hip_bf16.h>

// GraphSAGE 3-layer encoder for MI455X (gfx1250, wave32).
// Edge aggregation: f32 atomics (memory-bound phase, ~1.3 GB traffic).
// Dense GEMMs: v_wmma_f32_16x16x32_bf16, f32 accumulate, compile-time K/Kpad/Hout
// so the K-loop fully unrolls with no exec-mask predication.
// N = 50000 = 3125*16  -> exact 16-row tiling, no edge guards needed.

typedef __attribute__((ext_vector_type(16))) __bf16 v16bf;
typedef __attribute__((ext_vector_type(8)))  float  v8f;

#define NN 50000
#define NE 600000

static inline size_t align256(size_t x) { return (x + 255) & ~size_t(255); }

// ---------------------------------------------------------------- utilities

__global__ void fill_zero_f32(float* __restrict__ p, int n) {
  int i = blockIdx.x * blockDim.x + threadIdx.x;
  if (i < n) p[i] = 0.f;
}

__global__ void count_deg_kernel(const int* __restrict__ dst,
                                 float* __restrict__ deg, int E) {
  int e = blockIdx.x * blockDim.x + threadIdx.x;
  if (e < E) atomicAdd(&deg[dst[e]], 1.0f);
}

__global__ void rdeg_kernel(const float* __restrict__ deg,
                            float* __restrict__ rdeg, int n) {
  int i = blockIdx.x * blockDim.x + threadIdx.x;
  if (i < n) rdeg[i] = 1.0f / fmaxf(deg[i], 1.0f);
}

// Convert W (and optionally add W2) [Hout x K] f32 -> bf16 [Hout x Kpad], zero-padded in K.
__global__ void prep_weight_bf16(const float* __restrict__ W,
                                 const float* __restrict__ W2,
                                 __bf16* __restrict__ out,
                                 int Hout, int K, int Kpad) {
  int i = blockIdx.x * blockDim.x + threadIdx.x;
  int total = Hout * Kpad;
  if (i >= total) return;
  int r = i / Kpad, k = i - r * Kpad;
  float v = 0.f;
  if (k < K) {
    v = W[r * K + k];
    if (W2) v += W2[r * K + k];
  }
  out[i] = (__bf16)v;
}

// agg[dst] += h[src], float4-vectorized, one thread per (edge, 4-chunk).
__global__ void scatter_add_kernel(const float* __restrict__ h,
                                   const int* __restrict__ src,
                                   const int* __restrict__ dst,
                                   float* __restrict__ agg, int E, int D) {
  int tid = blockIdx.x * blockDim.x + threadIdx.x;
  int d4 = D >> 2;
  int e = tid / d4;
  if (e >= E) return;
  int d = (tid - e * d4) << 2;
  int s = src[e], t = dst[e];
  const float4 v = *reinterpret_cast<const float4*>(h + (size_t)s * D + d);
  float* p = agg + (size_t)t * D + d;
  atomicAdd(p + 0, v.x);
  atomicAdd(p + 1, v.y);
  atomicAdd(p + 2, v.z);
  atomicAdd(p + 3, v.w);
}

// ---------------------------------------------------------------- WMMA GEMM

// A fragment per ISA 7.12.2 (16-bit A 16x32): lane half 0 -> K {kb+0..7, kb+16..23},
// half 1 -> K {kb+8..15, kb+24..31}. Loads f32, scales, converts to bf16 in-register.
// K, kb compile-time -> the padding guards constant-fold (no runtime predication).
template <int K>
__device__ __forceinline__ void load_a_frag(const float* __restrict__ base,
                                            int kb, int half, float scale,
                                            v16bf& a) {
  const int k0 = kb + half * 8;
  const int k1 = kb + 16 + half * 8;
  if (kb < K) {  // both halves' first chunk in-bounds when kb < K (K multiple of 16)
#pragma unroll
    for (int j = 0; j < 8; ++j) a[j] = (__bf16)(base[k0 + j] * scale);
  } else {
#pragma unroll
    for (int j = 0; j < 8; ++j) a[j] = (__bf16)0.f;
  }
  if (kb + 16 < K) {
#pragma unroll
    for (int j = 0; j < 8; ++j) a[8 + j] = (__bf16)(base[k1 + j] * scale);
  } else {
#pragma unroll
    for (int j = 0; j < 8; ++j) a[8 + j] = (__bf16)0.f;
  }
}

// Y[N,HOUT] pre-activation:
//   (Agg * rdeg) @ Wl^T + Hin @ Wc^T + bias      (Wc = Wr + Wskip, pre-combined)
// One wave per 16x16 output tile; K swept in fully-unrolled 32-wide WMMA steps.
template <int K, int KPAD, int HOUT>
__global__ void __launch_bounds__(256)
sage_gemm_wmma(const float* __restrict__ Agg, const float* __restrict__ rdeg,
               const float* __restrict__ Hin,
               const __bf16* __restrict__ Wl, const __bf16* __restrict__ Wc,
               const float* __restrict__ bias,
               float* __restrict__ Y, int totalTiles) {
  constexpr int NCT = HOUT / 16;  // column tiles
  int wave = threadIdx.x >> 5;
  int lane = threadIdx.x & 31;
  int tile = blockIdx.x * (blockDim.x >> 5) + wave;
  if (tile >= totalTiles) return;  // whole-wave exit: EXEC all-ones for WMMA

  int rt = tile / NCT;
  int ct = tile - rt * NCT;
  int half = lane >> 4;
  int l16 = lane & 15;
  int row = rt * 16 + l16;  // A-matrix row for this lane (both halves share M)
  int col = ct * 16 + l16;  // B/C/D column for this lane

  float rd = rdeg[row];
  float bcol = bias[col];
  v8f acc;
#pragma unroll
  for (int e = 0; e < 8; ++e) acc[e] = bcol;  // bias folded into accumulator

  const float* pa = Agg + (size_t)row * K;
  const float* ph = Hin + (size_t)row * K;
  // B (32x16 bf16): lane reads W[col, kb + half*8 .. +7] and W[col, kb + 16 + half*8 .. +7]
  const __bf16* pwl = Wl + (size_t)col * KPAD + half * 8;
  const __bf16* pwc = Wc + (size_t)col * KPAD + half * 8;

#pragma unroll
  for (int kb = 0; kb < KPAD; kb += 32) {
    v16bf aA, aH;
    load_a_frag<K>(pa, kb, half, rd, aA);    // mean-normalized aggregate
    load_a_frag<K>(ph, kb, half, 1.0f, aH);  // node's own features
    v16bf bL, bC;
#pragma unroll
    for (int j = 0; j < 8; ++j) {
      bL[j] = pwl[kb + j]; bL[8 + j] = pwl[kb + 16 + j];
      bC[j] = pwc[kb + j]; bC[8 + j] = pwc[kb + 16 + j];
    }
    acc = __builtin_amdgcn_wmma_f32_16x16x32_bf16(false, aA, false, bL,
                                                  (short)0, acc, false, false);
    acc = __builtin_amdgcn_wmma_f32_16x16x32_bf16(false, aH, false, bC,
                                                  (short)0, acc, false, false);
  }

  // D layout: lane half 0 -> rows 0..7 (VGPR e), half 1 -> rows 8..15.
  float* py = Y + (size_t)(rt * 16 + half * 8) * HOUT + col;
#pragma unroll
  for (int e = 0; e < 8; ++e) py[(size_t)e * HOUT] = acc[e];
}

// ---------------------------------------------------------------- LayerNorm+ReLU

// One wave32 per node; D compile-time (128 or 64) so per-lane loops unroll.
template <int D>
__global__ void __launch_bounds__(256)
ln_relu_kernel(const float* __restrict__ Y, const float* __restrict__ g,
               const float* __restrict__ b, float* __restrict__ out, int N) {
  constexpr int PER = D / 32;  // elements per lane
  int wave = threadIdx.x >> 5;
  int lane = threadIdx.x & 31;
  int node = blockIdx.x * (blockDim.x >> 5) + wave;
  if (node >= N) return;
  const float* y = Y + (size_t)node * D;
  float vals[PER];
  float s = 0.f;
#pragma unroll
  for (int i = 0; i < PER; ++i) {
    vals[i] = y[lane + (i << 5)];
    s += vals[i];
  }
#pragma unroll
  for (int off = 16; off > 0; off >>= 1) s += __shfl_xor(s, off, 32);
  float mean = s / (float)D;
  float v = 0.f;
#pragma unroll
  for (int i = 0; i < PER; ++i) {
    float d = vals[i] - mean;
    v += d * d;
  }
#pragma unroll
  for (int off = 16; off > 0; off >>= 1) v += __shfl_xor(v, off, 32);
  float inv = rsqrtf(v / (float)D + 1e-5f);
  float* o = out + (size_t)node * D;
#pragma unroll
  for (int i = 0; i < PER; ++i) {
    int c = lane + (i << 5);
    float r = (vals[i] - mean) * inv * g[c] + b[c];
    o[c] = fmaxf(r, 0.f);
  }
}

// ---------------------------------------------------------------- launch

extern "C" void kernel_launch(void* const* d_in, const int* in_sizes, int n_in,
                              void* d_out, int out_size, void* d_ws, size_t ws_size,
                              hipStream_t stream) {
  (void)in_sizes; (void)n_in; (void)out_size; (void)ws_size;

  const float* x    = (const float*)d_in[0];
  const int*   ei   = (const int*)d_in[1];
  const int*   srcp = ei;        // edge_index[0]
  const int*   dstp = ei + NE;   // edge_index[1]
  const float* W1l = (const float*)d_in[2];
  const float* b1l = (const float*)d_in[3];
  const float* W1r = (const float*)d_in[4];
  const float* Ws1 = (const float*)d_in[5];
  const float* g1  = (const float*)d_in[6];
  const float* be1 = (const float*)d_in[7];
  const float* W2l = (const float*)d_in[8];
  const float* b2l = (const float*)d_in[9];
  const float* W2r = (const float*)d_in[10];
  const float* Ws2 = (const float*)d_in[11];
  const float* g2  = (const float*)d_in[12];
  const float* be2 = (const float*)d_in[13];
  const float* W3l = (const float*)d_in[14];
  const float* b3l = (const float*)d_in[15];
  const float* W3r = (const float*)d_in[16];
  const float* Ws3 = (const float*)d_in[17];
  const float* g3  = (const float*)d_in[18];
  const float* be3 = (const float*)d_in[19];
  float* outp = (float*)d_out;

  // workspace carve-up
  char* ws = (char*)d_ws;
  size_t off = 0;
  auto carve = [&](size_t bytes) -> void* {
    void* p = ws + off;
    off = align256(off + bytes);
    return p;
  };
  float*  deg  = (float*)carve((size_t)NN * 4);
  float*  rdg  = (float*)carve((size_t)NN * 4);
  float*  agg  = (float*)carve((size_t)NN * 128 * 4);
  float*  ybuf = (float*)carve((size_t)NN * 128 * 4);
  float*  h1   = (float*)carve((size_t)NN * 128 * 4);
  float*  h2   = (float*)carve((size_t)NN * 128 * 4);
  __bf16* wl1  = (__bf16*)carve((size_t)128 * 32 * 2);
  __bf16* wc1  = (__bf16*)carve((size_t)128 * 32 * 2);
  __bf16* wl2  = (__bf16*)carve((size_t)128 * 128 * 2);
  __bf16* wc2  = (__bf16*)carve((size_t)128 * 128 * 2);
  __bf16* wl3  = (__bf16*)carve((size_t)64 * 128 * 2);
  __bf16* wc3  = (__bf16*)carve((size_t)64 * 128 * 2);

  const int T = 256;

  // degrees + reciprocal (shared by all layers)
  fill_zero_f32<<<(NN + T - 1) / T, T, 0, stream>>>(deg, NN);
  count_deg_kernel<<<(NE + T - 1) / T, T, 0, stream>>>(dstp, deg, NE);
  rdeg_kernel<<<(NN + T - 1) / T, T, 0, stream>>>(deg, rdg, NN);

  // bf16 weight prep (Wc = Wr + Wskip pre-combined; layer1 K padded 16->32)
  prep_weight_bf16<<<(128 * 32 + T - 1) / T, T, 0, stream>>>(W1l, nullptr, wl1, 128, 16, 32);
  prep_weight_bf16<<<(128 * 32 + T - 1) / T, T, 0, stream>>>(W1r, Ws1, wc1, 128, 16, 32);
  prep_weight_bf16<<<(128 * 128 + T - 1) / T, T, 0, stream>>>(W2l, nullptr, wl2, 128, 128, 128);
  prep_weight_bf16<<<(128 * 128 + T - 1) / T, T, 0, stream>>>(W2r, Ws2, wc2, 128, 128, 128);
  prep_weight_bf16<<<(64 * 128 + T - 1) / T, T, 0, stream>>>(W3l, nullptr, wl3, 64, 128, 128);
  prep_weight_bf16<<<(64 * 128 + T - 1) / T, T, 0, stream>>>(W3r, Ws3, wc3, 64, 128, 128);

  // ---------------- layer 1: 16 -> 128
  fill_zero_f32<<<((NN * 16) + T - 1) / T, T, 0, stream>>>(agg, NN * 16);
  scatter_add_kernel<<<((NE * 4) + T - 1) / T, T, 0, stream>>>(x, srcp, dstp, agg, NE, 16);
  {
    int tiles = (NN / 16) * (128 / 16);
    sage_gemm_wmma<16, 32, 128><<<(tiles + 7) / 8, 256, 0, stream>>>(
        agg, rdg, x, wl1, wc1, b1l, ybuf, tiles);
  }
  ln_relu_kernel<128><<<(NN + 7) / 8, 256, 0, stream>>>(ybuf, g1, be1, h1, NN);

  // ---------------- layer 2: 128 -> 128
  fill_zero_f32<<<((NN * 128) + T - 1) / T, T, 0, stream>>>(agg, NN * 128);
  scatter_add_kernel<<<((NE * 32) + T - 1) / T, T, 0, stream>>>(h1, srcp, dstp, agg, NE, 128);
  {
    int tiles = (NN / 16) * (128 / 16);
    sage_gemm_wmma<128, 128, 128><<<(tiles + 7) / 8, 256, 0, stream>>>(
        agg, rdg, h1, wl2, wc2, b2l, ybuf, tiles);
  }
  ln_relu_kernel<128><<<(NN + 7) / 8, 256, 0, stream>>>(ybuf, g2, be2, h2, NN);

  // ---------------- layer 3: 128 -> 64
  fill_zero_f32<<<((NN * 128) + T - 1) / T, T, 0, stream>>>(agg, NN * 128);
  scatter_add_kernel<<<((NE * 32) + T - 1) / T, T, 0, stream>>>(h2, srcp, dstp, agg, NE, 128);
  {
    int tiles = (NN / 16) * (64 / 16);
    sage_gemm_wmma<128, 128, 64><<<(tiles + 7) / 8, 256, 0, stream>>>(
        agg, rdg, h2, wl3, wc3, b3l, ybuf, tiles);
  }
  ln_relu_kernel<64><<<(NN + 7) / 8, 256, 0, stream>>>(ybuf, g3, be3, outp, NN);
}
